// Attention_14207751815190
// MI455X (gfx1250) — matrix operand order repaired and checked by reading, not compiled, not run on hardware
//
#include <hip/hip_runtime.h>
#include <hip/hip_bf16.h>

// ---------------------------------------------------------------- constants
constexpr int CB   = 2;      // batch
constexpr int CS   = 1024;   // seq
constexpr int CHID = 1024;
constexpr int CNH  = 16;
constexpr int CNKV = 8;
constexpr int CG   = CNH / CNKV;     // 2
constexpr int CHD  = 128;
constexpr int CCL  = 3072;   // cache_len (fixed by harness)
constexpr int CMAXC= 4096;   // max cache
constexpr int CT   = CCL + CS;       // 4096 == CMAXC
constexpr int QSZ  = CNH * CHD;      // 2048
constexpr int KVSZ = CNKV * CHD;     // 1024
constexpr float QK_SCALE = 0.08838834764831845f;  // 128^-0.5
constexpr float LOG1E6   = 13.815510557964274f;   // ln(theta)

// ------------------------------------------------- async-to-LDS capability
#if defined(__has_builtin)
#  if __has_builtin(__builtin_amdgcn_global_load_async_to_lds_b128)
#    define HAVE_ASYNC_LDS 1
#  endif
#endif
#ifndef HAVE_ASYNC_LDS
#  define HAVE_ASYNC_LDS 0
#endif

typedef __attribute__((ext_vector_type(4))) int v4i;
#if HAVE_ASYNC_LDS
typedef __attribute__((address_space(1))) v4i g_v4i;   // global int4
typedef __attribute__((address_space(3))) v4i s_v4i;   // LDS int4
#endif

// 16B memory -> LDS copy; async (ASYNCcnt) when the toolchain exposes it.
__device__ __forceinline__ void copy_b128_to_lds(const unsigned short* g,
                                                 unsigned short* l) {
#if HAVE_ASYNC_LDS
  v4i* gv = (v4i*)g;   // pointee reinterpret (generic AS)
  v4i* lv = (v4i*)l;
  __builtin_amdgcn_global_load_async_to_lds_b128(
      (g_v4i*)gv, (s_v4i*)lv, /*offset=*/0, /*cpol=*/0);
#else
  *(uint4*)l = *(const uint4*)g;
#endif
}

__device__ __forceinline__ void copy_to_lds_wait() {
#if HAVE_ASYNC_LDS
#  if defined(__has_builtin) && __has_builtin(__builtin_amdgcn_s_wait_asynccnt)
  __builtin_amdgcn_s_wait_asynccnt(0);
#  else
  asm volatile("s_wait_asynccnt 0" ::: "memory");
#  endif
#endif
}

// ---------------------------------------------------------------- wmma types
typedef __attribute__((ext_vector_type(16))) __bf16 v16bf;
typedef __attribute__((ext_vector_type(8)))  float  v8f;

union FragU { unsigned int u[8]; v16bf b; };

__device__ __forceinline__ v8f vzero8() {
  v8f z = {0.f,0.f,0.f,0.f,0.f,0.f,0.f,0.f};
  return z;
}

__device__ __forceinline__ unsigned short f2bf(float f) {
  unsigned int u = __builtin_bit_cast(unsigned int, f);
  return (unsigned short)((u + 0x7FFFu + ((u >> 16) & 1u)) >> 16);
}
__device__ __forceinline__ unsigned int pack2(float lo, float hi) {
  return (unsigned int)f2bf(lo) | ((unsigned int)f2bf(hi) << 16);
}

__device__ __forceinline__ v8f wmma_bf16(v16bf a, v16bf b, v8f c) {
  return __builtin_amdgcn_wmma_f32_16x16x32_bf16(
      /*neg_a=*/false, a, /*neg_b=*/false, b,
      /*c_mod=*/(short)0, c, /*reuse_a=*/false, /*reuse_b=*/false);
}

// A fragment 16x32 bf16 (ISA 7.12.2): lane m = l&15, half h = l>>4 selects
// K-bank; VGPR v pairs K = ((v&4)<<2) + ((v&3)<<1) + 8h + {0,1}.
// `smem` points at row 0 of the tile (row-major MxK, stride in elements).
__device__ __forceinline__ v16bf load_frag_a(const unsigned short* smem,
                                             int stride, int lane) {
  int m = lane & 15, hh = (lane >> 4) & 1;
  const unsigned short* p = smem + m * stride + (hh << 3);
  FragU f;
#pragma unroll
  for (int v = 0; v < 8; ++v) {
    int kk = ((v & 4) << 2) + ((v & 3) << 1);
    f.u[v] = *(const unsigned int*)(p + kk);
  }
  return f.b;
}

// B fragment 32x16 bf16: lane column n = l&15; same K order as A:
// VGPR v pairs K = ((v&4)<<2) + ((v&3)<<1) + 8h + {0,1}.
// `smem` is the tile as [n][k] row-major (i.e. B-transposed, K contiguous).
__device__ __forceinline__ v16bf load_frag_b(const unsigned short* smem,
                                             int stride, int lane) {
  int n = lane & 15, hh = (lane >> 4) & 1;
  const unsigned short* p = smem + n * stride + (hh << 3);
  FragU f;
#pragma unroll
  for (int v = 0; v < 8; ++v) f.u[v] = *(const unsigned int*)(p + ((v & 4) << 2) + ((v & 3) << 1));
  return f.b;
}

// ---------------------------------------------------------------- kernel 1:
// fp32 caches -> bf16 K_all / V_all (layouts identical since CT == CMAXC).
__global__ void cache_cvt_kernel(const float* __restrict__ kc,
                                 const float* __restrict__ vc,
                                 unsigned short* __restrict__ Kall,
                                 unsigned short* __restrict__ Vall) {
  int i4 = blockIdx.x * blockDim.x + threadIdx.x;       // float4 index
  const int tot = CB * CCL * (KVSZ / 4);                // 1.57M
  if (i4 >= tot) return;
  int bt = i4 >> 8;               // (b,t) pair, 256 float4 per row
  int c4 = i4 & 255;
  int b  = bt / CCL;
  int t  = bt - b * CCL;
  size_t base4 = ((size_t)(b * CMAXC + t)) * 256 + c4;
  float4 dk = ((const float4*)kc)[base4];
  uint2 pk; pk.x = pack2(dk.x, dk.y); pk.y = pack2(dk.z, dk.w);
  ((uint2*)Kall)[base4] = pk;
  float4 dv = ((const float4*)vc)[base4];
  uint2 pv; pv.x = pack2(dv.x, dv.y); pv.y = pack2(dv.z, dv.w);
  ((uint2*)Vall)[base4] = pv;
}

// ---------------------------------------------------------------- kernel 2/5:
// C[M,N] (fp32) = A[M,K] * B[N,K]^T via bf16 WMMA. 64x64 tile / 256 threads.
constexpr int GTS = 64, GKS = 32, GLS = 40;   // tile, k-step, LDS row stride

template <bool A_BF16>
__global__ void __launch_bounds__(256)
gemm_wmma_kernel(const void* __restrict__ Ain, const float* __restrict__ Bn,
                 float* __restrict__ C, int M, int N, int K) {
  __shared__ unsigned short As[GTS * GLS];
  __shared__ unsigned short Bs[GTS * GLS];
  int tid = threadIdx.x, lane = tid & 31, w = tid >> 5;
  int bm = blockIdx.y, bn = blockIdx.x;
  int mi = w >> 1, ni0 = (w & 1) * 2;       // wave -> 1 m-subtile, 2 n-subtiles
  v8f acc0 = vzero8(), acc1 = vzero8();
  const float*          Af = (const float*)Ain;
  const unsigned short* Ab = (const unsigned short*)Ain;

  for (int k0 = 0; k0 < K; k0 += GKS) {
    __syncthreads();
    if (A_BF16) {
      // pure bf16 copy -> async-to-LDS datapath when available
      int r = tid >> 2, c = (tid & 3) * 8;                  // 64 rows x 32 bf16
      const unsigned short* src = Ab + (size_t)(bm * GTS + r) * K + k0 + c;
      copy_b128_to_lds(src, &As[r * GLS + c]);
      __builtin_prefetch(src + GKS, 0, 1);
    } else {
#pragma unroll
      for (int i = 0; i < 2; ++i) {
        int f = tid + i * 256, r = f >> 3, c = (f & 7) * 4;
        const float* src = Af + (size_t)(bm * GTS + r) * K + k0 + c;
        float4 d = *(const float4*)src;
        uint2 pk; pk.x = pack2(d.x, d.y); pk.y = pack2(d.z, d.w);
        *(uint2*)&As[r * GLS + c] = pk;
        __builtin_prefetch(src + GKS, 0, 1);
      }
    }
#pragma unroll
    for (int i = 0; i < 2; ++i) {
      int f = tid + i * 256, r = f >> 3, c = (f & 7) * 4;
      const float* src = Bn + (size_t)(bn * GTS + r) * K + k0 + c;
      float4 d = *(const float4*)src;
      uint2 pk; pk.x = pack2(d.x, d.y); pk.y = pack2(d.z, d.w);
      *(uint2*)&Bs[r * GLS + c] = pk;
      __builtin_prefetch(src + GKS, 0, 1);
    }
    if (A_BF16) copy_to_lds_wait();
    __syncthreads();
    v16bf a  = load_frag_a(As + (mi * 16) * GLS, GLS, lane);
    v16bf b0 = load_frag_b(Bs + (ni0 * 16) * GLS, GLS, lane);
    v16bf b1 = load_frag_b(Bs + ((ni0 + 1) * 16) * GLS, GLS, lane);
    acc0 = wmma_bf16(a, b0, acc0);
    acc1 = wmma_bf16(a, b1, acc1);
  }

  int n = lane & 15, hh = lane >> 4;
#pragma unroll
  for (int r = 0; r < 8; ++r) {
    size_t row = (size_t)(bm * GTS + mi * 16 + r + 8 * hh) * N;
    C[row + bn * GTS + ni0 * 16 + n]        = acc0[r];
    C[row + bn * GTS + (ni0 + 1) * 16 + n]  = acc1[r];
  }
}

// ---------------------------------------------------------------- kernel 3:
// per-(b,s,head): RMSNorm + RoPE; q -> bf16 (scaled), k/v -> K_all/V_all tail.
__global__ void __launch_bounds__(128)
prep_kernel(const float* __restrict__ qkv, const float* __restrict__ qnw,
            const float* __restrict__ knw, unsigned short* __restrict__ Qb,
            unsigned short* __restrict__ Kall, unsigned short* __restrict__ Vall) {
  __shared__ float xs[CHD];
  __shared__ float red[4];
  int d  = threadIdx.x;
  int bx = blockIdx.x;
  int hh = bx & 31;                 // 16 q + 8 k + 8 v slots
  int s  = (bx >> 5) & (CS - 1);
  int b  = bx >> 15;
  const float* row = qkv + (size_t)(b * CS + s) * (QSZ + 2 * KVSZ);

  if (hh >= 24) {                                   // v head: plain bf16 copy
    int kvh = hh - 24;
    float val = row[QSZ + KVSZ + kvh * CHD + d];
    Vall[((size_t)(b * CT + CCL + s) * CNKV + kvh) * CHD + d] = f2bf(val);
    return;
  }
  bool is_q = hh < 16;
  int  head = is_q ? hh : hh - 16;
  int  off  = is_q ? head * CHD : QSZ + head * CHD;
  const float* wn = is_q ? qnw : knw;

  float x  = row[off + d];
  float ss = x * x;
#pragma unroll
  for (int o = 16; o; o >>= 1) ss += __shfl_xor(ss, o, 32);
  if ((d & 31) == 0) red[d >> 5] = ss;
  __syncthreads();
  float tot = red[0] + red[1] + red[2] + red[3];
  float rs  = rsqrtf(tot * (1.0f / CHD) + 1e-6f);
  float xn  = x * rs * wn[d];
  xs[d] = xn;
  __syncthreads();
  float partner = xs[d ^ 64];

  int   i    = d & 63;
  float invf = __expf(-(float)i * (LOG1E6 / 64.0f));
  float ang  = (float)(CCL + s) * invf;
  float sn, cs;
  __sincosf(ang, &sn, &cs);
  float out = xn * cs + (d < 64 ? -partner : partner) * sn;

  if (is_q)
    Qb[((size_t)(b * CS + s) * CNH + head) * CHD + d] = f2bf(out * QK_SCALE);
  else
    Kall[((size_t)(b * CT + CCL + s) * CNKV + head) * CHD + d] = f2bf(out);
}

// ---------------------------------------------------------------- kernel 4:
// flash attention: 4 waves/WG, each owns 16 queries of one (b, q-head);
// 32-key tiles of K (row-major) and V (transposed) staged in LDS as bf16.
constexpr int AKT  = 32;          // keys per tile
constexpr int KSTR = 136;         // Ks row stride (HD + 8), 16B-aligned rows
constexpr int VSTR = 40;          // Vt row stride (32 + 8)
constexpr int PSTR = 36;          // P  row stride (32 + 4)

__global__ void __launch_bounds__(128)
attn_kernel(const unsigned short* __restrict__ Qb,
            const unsigned short* __restrict__ Kall,
            const unsigned short* __restrict__ Vall,
            unsigned short* __restrict__ Ctx) {
  __shared__ unsigned short Ks[AKT * KSTR];      // [key][hd]
  __shared__ unsigned short Vt[CHD * VSTR];      // [hd][key]
  __shared__ unsigned short Ps[4][16 * PSTR];    // per-wave P tile [m][key]

  int tid = threadIdx.x, lane = tid & 31, w = tid >> 5;
  int bx = blockIdx.x;
  int qb = bx & 15;                 // 64-query block
  int h  = (bx >> 4) & 15;          // q head
  int b  = bx >> 8;
  int kv = h >> 1;                  // G = 2
  int s_base = qb * 64 + w * 16;
  int m = lane & 15, hh = lane >> 4, n = lane & 15;

  // preload the wave's Q fragments (4 k-steps of 32 over HD=128)
  const unsigned short* qrow =
      Qb + ((size_t)(b * CS + s_base + m) * CNH + h) * CHD + (hh << 3);
  v16bf qf[4];
#pragma unroll
  for (int kk = 0; kk < 4; ++kk) {
    FragU f;
#pragma unroll
    for (int v = 0; v < 8; ++v) {
      int off = kk * 32 + ((v & 4) << 2) + ((v & 3) << 1);
      f.u[v] = *(const unsigned int*)(qrow + off);
    }
    qf[kk] = f.b;
  }

  v8f oacc[8];
#pragma unroll
  for (int j = 0; j < 8; ++j) oacc[j] = vzero8();
  float mrow[8], lrow[8];
#pragma unroll
  for (int r = 0; r < 8; ++r) { mrow[r] = -1e30f; lrow[r] = 0.f; }

  const int t_end = CCL + qb * 64 + 64;          // uniform across the block
  for (int t0 = 0; t0 < t_end; t0 += AKT) {
    { // cooperative staging: 128 threads, 1 key row per 4 threads
      int r = tid >> 2, dbase = (tid & 3) * 32;
      const unsigned short* krow =
          Kall + ((size_t)(b * CT + t0 + r) * CNKV + kv) * CHD;
      const unsigned short* vrow =
          Vall + ((size_t)(b * CT + t0 + r) * CNKV + kv) * CHD;
#pragma unroll
      for (int j = 0; j < 4; ++j) {
        int d = dbase + j * 8;
        // K tile: straight bf16 copy -> async-to-LDS datapath when available
        copy_b128_to_lds(krow + d, &Ks[r * KSTR + d]);
        // V tile: needs a transpose, manual path
        union { uint4 q; unsigned short s[8]; } uv;
        uv.q = *(const uint4*)(vrow + d);
#pragma unroll
        for (int jj = 0; jj < 8; ++jj) Vt[(d + jj) * VSTR + r] = uv.s[jj];
      }
      copy_to_lds_wait();
    }
    __syncthreads();

    // scores S[16 x 32] = Q * K^T (q already scaled by HD^-0.5)
    v8f s0 = vzero8(), s1 = vzero8();
#pragma unroll
    for (int kk = 0; kk < 4; ++kk) {
      v16bf b0 = load_frag_b(Ks + kk * 32, KSTR, lane);
      v16bf b1 = load_frag_b(Ks + 16 * KSTR + kk * 32, KSTR, lane);
      s0 = wmma_bf16(qf[kk], b0, s0);
      s1 = wmma_bf16(qf[kk], b1, s1);
    }

    // causal mask + online softmax (row stats live per C-layout VGPR row)
    float p0[8], p1[8];
#pragma unroll
    for (int r = 0; r < 8; ++r) {
      int pos = CCL + s_base + r + 8 * hh;
      float v0 = s0[r]; if (t0 + n      > pos) v0 = -1e30f;
      float v1 = s1[r]; if (t0 + 16 + n > pos) v1 = -1e30f;
      float mx = fmaxf(v0, v1);
#pragma unroll
      for (int o = 8; o; o >>= 1) mx = fmaxf(mx, __shfl_xor(mx, o, 32));
      float mnew  = fmaxf(mrow[r], mx);
      float scale = __expf(mrow[r] - mnew);
      float e0 = __expf(v0 - mnew), e1 = __expf(v1 - mnew);
      float sum = e0 + e1;
#pragma unroll
      for (int o = 8; o; o >>= 1) sum += __shfl_xor(sum, o, 32);
      lrow[r] = lrow[r] * scale + sum;
      mrow[r] = mnew;
      p0[r] = e0; p1[r] = e1;
#pragma unroll
      for (int j = 0; j < 8; ++j) oacc[j][r] *= scale;
    }

    // P: C-layout -> A-layout via per-wave LDS round trip
    unsigned short* pw = Ps[w];
#pragma unroll
    for (int r = 0; r < 8; ++r) {
      int mm = r + 8 * hh;
      pw[mm * PSTR + n]      = f2bf(p0[r]);
      pw[mm * PSTR + 16 + n] = f2bf(p1[r]);
    }
    asm volatile("s_wait_dscnt 0" ::: "memory");   // same-wave cross-lane hop
    v16bf ap = load_frag_a(pw, PSTR, lane);

    // O[16 x 128] += P[16 x 32] * V[32 x 128]
#pragma unroll
    for (int j = 0; j < 8; ++j) {
      v16bf bv = load_frag_b(Vt + (j * 16) * VSTR, VSTR, lane);
      oacc[j] = wmma_bf16(ap, bv, oacc[j]);
    }
    __syncthreads();
  }

  // normalize + write ctx (bf16, [B,S,NH,HD])
#pragma unroll
  for (int r = 0; r < 8; ++r) {
    float inv = 1.0f / lrow[r];
    int sq = s_base + r + 8 * hh;
    unsigned short* crow = Ctx + ((size_t)(b * CS + sq) * CNH + h) * CHD;
#pragma unroll
    for (int j = 0; j < 8; ++j) crow[j * 16 + n] = f2bf(oacc[j][r] * inv);
  }
}

// ---------------------------------------------------------------- launcher
extern "C" void kernel_launch(void* const* d_in, const int* in_sizes, int n_in,
                              void* d_out, int out_size, void* d_ws,
                              size_t ws_size, hipStream_t stream) {
  const float* hidden = (const float*)d_in[0];   // [B,S,HID]
  const float* qkvw   = (const float*)d_in[1];   // [4096,1024]
  const float* qnw    = (const float*)d_in[2];   // [128]
  const float* knw    = (const float*)d_in[3];   // [128]
  const float* ow     = (const float*)d_in[4];   // [1024,2048]
  const float* kcache = (const float*)d_in[5];   // [B,4096,8,128]
  const float* vcache = (const float*)d_in[6];   // [B,4096,8,128]
  float* out = (float*)d_out;                    // [B,S,HID]

  char* ws = (char*)d_ws;
  float*          qkv  = (float*)ws;                                // 32 MB
  unsigned short* Qb   = (unsigned short*)(ws + (33554432));        //  8 MB
  unsigned short* Kall = (unsigned short*)(ws + (33554432 + 8388608));        // 16 MB
  unsigned short* Vall = (unsigned short*)(ws + (33554432 + 8388608 + 16777216)); // 16 MB
  unsigned short* Ctx  = (unsigned short*)(ws + (33554432 + 8388608 + 33554432)); //  8 MB

  // 1) bf16-convert the existing cache region (t < 3072)
  cache_cvt_kernel<<<6144, 256, 0, stream>>>(kcache, vcache, Kall, Vall);

  // 2) QKV projection: [2048 x 4096] = hidden[2048 x 1024] * Wqkv^T
  gemm_wmma_kernel<false><<<dim3(4096 / GTS, 2048 / GTS), 256, 0, stream>>>(
      hidden, qkvw, qkv, CB * CS, QSZ + 2 * KVSZ, CHID);

  // 3) RMSNorm + RoPE + cache tail append
  prep_kernel<<<CB * CS * 32, 128, 0, stream>>>(qkv, qnw, knw, Qb, Kall, Vall);

  // 4) attention -> bf16 ctx
  attn_kernel<<<CB * CNH * (CS / 64), 128, 0, stream>>>(Qb, Kall, Vall, Ctx);

  // 5) output projection: out[2048 x 1024] = ctx[2048 x 2048] * Wo^T
  gemm_wmma_kernel<true><<<dim3(1024 / GTS, 2048 / GTS), 256, 0, stream>>>(
      Ctx, ow, out, CB * CS, CHID, QSZ);
}
